// EnhancedWorldModel_86431921864747
// MI455X (gfx1250) — compile-verified, matrix operand order checked by reading, not run on hardware
//
#include <hip/hip_runtime.h>
#include <hip/hip_bf16.h>
#include <math.h>

typedef __attribute__((ext_vector_type(16))) _Float16 v16h;
typedef __attribute__((ext_vector_type(8)))  _Float16 v8h;
typedef __attribute__((ext_vector_type(4)))  _Float16 v4h;
typedef __attribute__((ext_vector_type(8)))  float    v8f;
typedef __attribute__((ext_vector_type(4)))  float    v4f;
typedef __attribute__((ext_vector_type(4)))  unsigned int u32x4;
typedef __attribute__((ext_vector_type(8)))  int      i32x8;
typedef __attribute__((ext_vector_type(4)))  int      i32x4;

#define D        512
#define NHEADS   8
#define DH       64
#define NTOK     7
#define BATCH    64
#define Q        448        /* BATCH*NTOK */
#define HID      2048
#define MEMROWS  100000
#define TOPK     5
#define NLAYERS  8
#define LN_EPS   1e-5f
#define KP       32         /* mem rows per KNN panel */
#define NPANEL   32         /* panels per chunk -> chunk = 1024 rows */
#define NCHUNKS  98         /* ceil(100000/1024) */
#define TDM_ROWS 16         /* rows per TDM copy tile (100000 % 16 == 0) */

__device__ __forceinline__ float gelu_f(float x) {
  float x3 = x * x * x;
  return 0.5f * x * (1.0f + tanhf(0.7978845608028654f * (x + 0.044715f * x3)));
}

/* 16x16x32 f16 WMMA fragment load: per ISA 7.12.2 each lane's 16 halves are
   two contiguous 8-half runs along K:  [g8, g8+8)  and  [16+g8, 16+g8+8)
   with g8 = 8*(lane>=16).  => exactly two 16B (b128) loads. */
__device__ __forceinline__ v16h frag_load(const _Float16* p, int lane) {
  const v8h* q = (const v8h*)(p + ((lane & 16) ? 8 : 0));
  v8h lo = q[0];
  v8h hi = q[2];            /* +16 halves */
  return __builtin_shufflevector(lo, hi, 0, 1, 2, 3, 4, 5, 6, 7,
                                 8, 9, 10, 11, 12, 13, 14, 15);
}

/* ---------------- generic f32-in/f32-out GEMM via f16 WMMA ----------------
   C[M,N] = act(A[M,K] @ B[K,N] + bias) ; optional C = resid + gate*(...)
   M % 32 == 0, N % 64 == 0, K % 64 == 0 (true for every layer here).
   K staged 64-wide -> two back-to-back v_wmma per barrier pair.            */
#define KC 64
__global__ __launch_bounds__(256)
void gemm_wmma(const float* __restrict__ A, const float* __restrict__ Bw,
               const float* __restrict__ bias, const float* __restrict__ resid,
               const float* __restrict__ gate, float* __restrict__ C,
               int M, int N, int K, int act)
{
  __shared__ __align__(16) _Float16 As[32][KC + 8];    /* 144B row stride */
  __shared__ __align__(16) _Float16 Bst[64][KC + 8];   /* transposed: [col][k] */
  int tid  = threadIdx.x;
  int lane = tid & 31;
  int w    = tid >> 5;
  int wr   = (w >> 2) << 4;          /* 0 or 16   */
  int wc   = (w & 3) << 4;           /* 0..48     */
  int bm   = blockIdx.y << 5;
  int bn   = blockIdx.x << 6;
  int ar   = wr + (lane & 15);
  int bc   = wc + (lane & 15);
  v8f c = {};
  for (int k0 = 0; k0 < K; k0 += KC) {
#pragma unroll
    for (int i = 0; i < 2; ++i) {   /* A tile 32x64 */
      int e  = tid + (i << 8);
      int r  = e >> 4;
      int kq = (e & 15) << 2;
      v4f t = *(const v4f*)&A[(size_t)(bm + r) * K + k0 + kq];
      *(v4h*)&As[r][kq] = __builtin_convertvector(t, v4h);
    }
#pragma unroll
    for (int i = 0; i < 4; ++i) {   /* B tile 64x64 -> transposed LDS */
      int e  = tid + (i << 8);
      int cc = e & 63;
      int r4 = (e >> 6) << 2;
      v4f t;
      t[0] = Bw[(size_t)(k0 + r4 + 0) * N + bn + cc];
      t[1] = Bw[(size_t)(k0 + r4 + 1) * N + bn + cc];
      t[2] = Bw[(size_t)(k0 + r4 + 2) * N + bn + cc];
      t[3] = Bw[(size_t)(k0 + r4 + 3) * N + bn + cc];
      *(v4h*)&Bst[cc][r4] = __builtin_convertvector(t, v4h);
    }
    __syncthreads();
#pragma unroll
    for (int kk = 0; kk < KC; kk += 32) {
      v16h a = frag_load(&As[ar][kk], lane);
      v16h b = frag_load(&Bst[bc][kk], lane);
      c = __builtin_amdgcn_wmma_f32_16x16x32_f16(false, a, false, b, (short)0, c,
                                                 false, false);
    }
    __syncthreads();
  }
  int gm0 = bm + wr + ((lane & 16) ? 8 : 0);
  int gn  = bn + wc + (lane & 15);
#pragma unroll
  for (int i = 0; i < 8; ++i) {
    int gm  = gm0 + i;
    float v = c[i];
    if (bias) v += bias[gn];
    if (act == 1)      v = gelu_f(v);
    else if (act == 2) v = 1.0f / (1.0f + __expf(-v));
    if (resid) {
      float g = gate ? gate[0] : 1.0f;
      v = resid[(size_t)gm * N + gn] + g * v;
    }
    C[(size_t)gm * N + gn] = v;
  }
}

/* ---------------- LayerNorm over rows (optionally + gelu) ---------------- */
__global__ __launch_bounds__(256)
void ln_rows(const float* __restrict__ X, const float* __restrict__ g,
             const float* __restrict__ b, float* __restrict__ Y, int C, int dogelu)
{
  __shared__ float s1[256], s2[256];
  int row = blockIdx.x;
  const float* xr = X + (size_t)row * C;
  float sum = 0.f, sq = 0.f;
  for (int i = threadIdx.x; i < C; i += 256) { float v = xr[i]; sum += v; sq += v * v; }
  s1[threadIdx.x] = sum; s2[threadIdx.x] = sq;
  __syncthreads();
  for (int s = 128; s > 0; s >>= 1) {
    if (threadIdx.x < s) { s1[threadIdx.x] += s1[threadIdx.x + s];
                           s2[threadIdx.x] += s2[threadIdx.x + s]; }
    __syncthreads();
  }
  float mu   = s1[0] / C;
  float var  = s2[0] / C - mu * mu;
  float rstd = rsqrtf(var + LN_EPS);
  float* yr = Y + (size_t)row * C;
  for (int i = threadIdx.x; i < C; i += 256) {
    float v = (xr[i] - mu) * rstd * g[i] + b[i];
    if (dogelu) v = gelu_f(v);
    yr[i] = v;
  }
}

/* ---------------- tiny 7x7 attention with relative position bias ---------------- */
__global__ __launch_bounds__(64)
void attn_small(const float* __restrict__ qkv, const float* __restrict__ rel,
                float* __restrict__ O)
{
  int b = blockIdx.x >> 3;
  int h = blockIdx.x & 7;
  __shared__ float s[NTOK][NTOK];
  __shared__ float qs[NTOK][DH], ks[NTOK][DH], vs[NTOK][DH];
  int t = threadIdx.x;
  for (int i = t; i < NTOK * DH; i += 64) {
    int n = i >> 6, d = i & 63;
    size_t base = (size_t)(b * NTOK + n) * (3 * D) + h * DH + d;
    qs[n][d] = qkv[base];
    ks[n][d] = qkv[base + D];
    vs[n][d] = qkv[base + 2 * D];
  }
  __syncthreads();
  if (t < NTOK * NTOK) {
    int n = t / NTOK, m = t % NTOK;
    float acc = 0.f;
    for (int d = 0; d < DH; ++d) acc += qs[n][d] * ks[m][d];
    s[n][m] = acc * 0.125f + rel[(n - m + NTOK - 1) * NHEADS + h];
  }
  __syncthreads();
  if (t < NTOK) {
    float mx = -1e30f;
    for (int m = 0; m < NTOK; ++m) mx = fmaxf(mx, s[t][m]);
    float sum = 0.f;
    for (int m = 0; m < NTOK; ++m) { float e = __expf(s[t][m] - mx); s[t][m] = e; sum += e; }
    float inv = 1.f / sum;
    for (int m = 0; m < NTOK; ++m) s[t][m] *= inv;
  }
  __syncthreads();
  for (int i = t; i < NTOK * DH; i += 64) {
    int n = i >> 6, d = i & 63;
    float acc = 0.f;
    for (int m = 0; m < NTOK; ++m) acc += s[n][m] * vs[m][d];
    O[(size_t)(b * NTOK + n) * D + h * DH + d] = acc;
  }
}

/* ---------------- per-row 1/max(||x||,1e-8) ---------------- */
__global__ __launch_bounds__(128)
void row_inv_norm(const float* __restrict__ X, float* __restrict__ inv, int C)
{
  __shared__ float s1[128];
  int row = blockIdx.x;
  const float* xr = X + (size_t)row * C;
  float sq = 0.f;
  for (int i = threadIdx.x; i < C; i += 128) { float v = xr[i]; sq += v * v; }
  s1[threadIdx.x] = sq;
  __syncthreads();
  for (int s = 64; s > 0; s >>= 1) {
    if (threadIdx.x < s) s1[threadIdx.x] += s1[threadIdx.x + s];
    __syncthreads();
  }
  if (threadIdx.x == 0) inv[row] = 1.0f / fmaxf(sqrtf(s1[0]), 1e-8f);
}

/* normalized query rows -> f16 for the WMMA similarity GEMM */
__global__ __launch_bounds__(256)
void scale_to_h(const float* __restrict__ X, const float* __restrict__ inv,
                _Float16* __restrict__ Y, int total)
{
  int i = blockIdx.x * 256 + threadIdx.x;
  if (i < total) Y[i] = (_Float16)(X[i] * inv[i >> 9]);   /* C = 512 */
}

/* ---------------- fused cosine-sim GEMM + running top-5 ----------------- */
__global__ __launch_bounds__(256)
void knn_topk(const _Float16* __restrict__ qnh, const float* __restrict__ mkeys,
              const float* __restrict__ minv, float* __restrict__ pval,
              int* __restrict__ pidx)
{
  __shared__ __align__(16) _Float16 Bs[KP][D];  /* 32 KB normalized keys (f16) */
  __shared__ float simS[64][33];                /* sim tile scratch */
  __shared__ float tv[Q][TOPK];                 /* running top-5 values  */
  __shared__ int   ti[Q][TOPK];                 /* running top-5 indices */
  int tid = threadIdx.x, lane = tid & 31, w = tid >> 5;
  int cbase = blockIdx.x * (KP * NPANEL);
  for (int i = tid; i < Q * TOPK; i += 256) {
    ((float*)tv)[i] = -3.0e38f; ((int*)ti)[i] = 0;
  }
  __syncthreads();
  int wr = (w >> 1) << 4;  /* 0..48 : 64 query rows per pass */
  int wc = (w & 1) << 4;   /* 0/16  : 32 memory rows per panel */
  for (int p = 0; p < NPANEL; ++p) {
    int mbase = cbase + p * KP;
    if (p + 1 < NPANEL && mbase + KP < MEMROWS)
      __builtin_prefetch(&mkeys[(size_t)(mbase + KP) * D + (tid << 4)], 0, 3);
    /* stage panel: f32x4 global load -> scale -> f16x4 LDS store */
#pragma unroll
    for (int i = 0; i < 16; ++i) {
      int e = tid + (i << 8);          /* 4096 f32x4 slots */
      int r = e >> 7;
      int kq = (e & 127) << 2;
      int mrow = mbase + r;
      v4f t = {0.f, 0.f, 0.f, 0.f};
      if (mrow < MEMROWS) {
        t = *(const v4f*)&mkeys[(size_t)mrow * D + kq];
        t = t * minv[mrow];
      }
      *(v4h*)&Bs[r][kq] = __builtin_convertvector(t, v4h);
    }
    __syncthreads();
    for (int qt = 0; qt < Q; qt += 64) {
      v8f c = {};
      int arow = qt + wr + (lane & 15);
      int brow = wc + (lane & 15);
      const _Float16* aptr = qnh + (size_t)arow * D;
      for (int k0 = 0; k0 < D; k0 += 32) {
        v16h a = frag_load(aptr + k0, lane);
        v16h b = frag_load(&Bs[brow][k0], lane);
        c = __builtin_amdgcn_wmma_f32_16x16x32_f16(false, a, false, b, (short)0, c,
                                                   false, false);
      }
      int sr0 = wr + ((lane & 16) ? 8 : 0);
      int sc  = wc + (lane & 15);
#pragma unroll
      for (int i = 0; i < 8; ++i) simS[sr0 + i][sc] = c[i];
      __syncthreads();
      if (tid < 64) {                  /* each thread owns one query row */
        int q = qt + tid;
        for (int cc = 0; cc < KP; ++cc) {
          int mrow = mbase + cc;
          if (mrow >= MEMROWS) break;
          float v = simS[tid][cc];
          if (v > tv[q][TOPK - 1]) {
            int j = TOPK - 1;
            while (j > 0 && tv[q][j - 1] < v) {
              tv[q][j] = tv[q][j - 1]; ti[q][j] = ti[q][j - 1]; --j;
            }
            tv[q][j] = v; ti[q][j] = mrow;
          }
        }
      }
      __syncthreads();
    }
  }
  for (int i = tid; i < Q * TOPK; i += 256) {
    pval[(size_t)blockIdx.x * Q * TOPK + i] = ((float*)tv)[i];
    pidx[(size_t)blockIdx.x * Q * TOPK + i] = ((int*)ti)[i];
  }
}

/* ---------------- merge partial top-5s + 5-way memory attention ---------------- */
__global__ __launch_bounds__(256)
void mem_attend(const float* __restrict__ pval, const int* __restrict__ pidx,
                int nchunks, const float* __restrict__ qf,
                const float* __restrict__ mkeys, const float* __restrict__ mvals,
                float* __restrict__ x)
{
  __shared__ int   sel[TOPK];
  __shared__ float wts[TOPK];
  __shared__ float red[256];
  int q = blockIdx.x, t = threadIdx.x;
  if (t == 0) {
    float bv[TOPK]; int bi[TOPK];
    for (int j = 0; j < TOPK; ++j) { bv[j] = -3.0e38f; bi[j] = 0; }
    for (int ch = 0; ch < nchunks; ++ch) {
      const float* pv = pval + ((size_t)ch * Q + q) * TOPK;
      const int*   pi = pidx + ((size_t)ch * Q + q) * TOPK;
      for (int j = 0; j < TOPK; ++j) {
        float v = pv[j]; int id = pi[j];
        if (v > bv[TOPK - 1]) {
          int jj = TOPK - 1;
          while (jj > 0 && bv[jj - 1] < v) { bv[jj] = bv[jj - 1]; bi[jj] = bi[jj - 1]; --jj; }
          bv[jj] = v; bi[jj] = id;
        }
      }
    }
    for (int j = 0; j < TOPK; ++j) sel[j] = bi[j];
  }
  __syncthreads();
  float dots[TOPK];
  const float* qr = qf + (size_t)q * D;
  for (int j = 0; j < TOPK; ++j) {
    const float* kr = mkeys + (size_t)sel[j] * D;
    float acc = 0.f;
    for (int i = t; i < D; i += 256) acc += qr[i] * kr[i];
    red[t] = acc; __syncthreads();
    for (int s = 128; s > 0; s >>= 1) {
      if (t < s) red[t] += red[t + s];
      __syncthreads();
    }
    dots[j] = red[0] * 0.04419417382415922f;   /* 1/sqrt(512) */
    __syncthreads();
  }
  if (t == 0) {
    float mx = -1e30f;
    for (int j = 0; j < TOPK; ++j) mx = fmaxf(mx, dots[j]);
    float sum = 0.f, e[TOPK];
    for (int j = 0; j < TOPK; ++j) { e[j] = __expf(dots[j] - mx); sum += e[j]; }
    for (int j = 0; j < TOPK; ++j) wts[j] = e[j] / sum;
  }
  __syncthreads();
  for (int i = t; i < D; i += 256) {
    float acc = 0.f;
    for (int j = 0; j < TOPK; ++j) acc += wts[j] * mvals[(size_t)sel[j] * D + i];
    x[(size_t)q * D + i] += acc;
  }
}

/* ---------------- TDM tile copy: global -> LDS -> global ----------------
   Exercises the Tensor Data Mover for the 2x205MB mk/mv write-back.
   Descriptors per CDNA5 ISA 8.3/8.4: count=1, type=2, data_size=4B,
   2D tensor (dim0=512, dim1=remaining rows), tile 512 x TDM_ROWS.        */
#if __has_builtin(__builtin_amdgcn_tensor_load_to_lds) && \
    __has_builtin(__builtin_amdgcn_tensor_store_from_lds)
#define HAVE_TDM 1
#pragma message("CDNA5 probe: tensor_load_to_lds builtins AVAILABLE -> TDM path enabled")
#else
#pragma message("CDNA5 probe: tensor_load_to_lds builtins MISSING -> fallback copy path")
#endif
#if __has_builtin(__builtin_amdgcn_s_wait_tensorcnt)
#pragma message("CDNA5 probe: s_wait_tensorcnt builtin AVAILABLE")
#endif

__global__ __launch_bounds__(32)
void tdm_copy_rows(const float* __restrict__ src, float* __restrict__ dst, int rows)
{
#ifdef HAVE_TDM
  __shared__ __align__(16) float tile[TDM_ROWS * D];
  /* keep the LDS allocation alive: TDM writes to LDS offset 0 */
  asm volatile("" ::"v"(&tile[0]) : "memory");
  int r0 = blockIdx.x * TDM_ROWS;
  unsigned remain = (unsigned)(rows - r0);
  unsigned long long ga =
      (unsigned long long)(const void*)src + (unsigned long long)r0 * D * 4ull;
  u32x4 g0; i32x8 g1; i32x4 g2 = {0, 0, 0, 0}, g3 = {0, 0, 0, 0};
  g0[0] = 1u;                                          /* count=1           */
  g0[1] = 0u;                                          /* lds_addr = 0      */
  g0[2] = (unsigned)(ga & 0xffffffffu);
  g0[3] = (unsigned)((ga >> 32) & 0x01ffffffu) | (2u << 30);  /* type=2     */
  g1[0] = 0x00020000;                                  /* data_size = 4B    */
  g1[1] = (int)(512u << 16);                           /* tensor_dim0 = 512 */
  g1[2] = (int)((remain & 0xffffu) << 16);             /* tensor_dim1 lo    */
  g1[3] = (int)(((remain >> 16) & 0xffffu) | (512u << 16)); /* hi|tile_dim0 */
  g1[4] = TDM_ROWS;                                    /* tile_dim1, dim2=0 */
  g1[5] = 512;                                         /* dim0_stride low32 */
  g1[6] = 0;
  g1[7] = 0;
#if defined(__clang_major__) && __clang_major__ >= 23
  i32x8 gz = {};
  __builtin_amdgcn_tensor_load_to_lds(g0, g1, g2, g3, gz, 0);
#else
  __builtin_amdgcn_tensor_load_to_lds(g0, g1, g2, g3, 0);
#endif
#if __has_builtin(__builtin_amdgcn_s_wait_tensorcnt)
  __builtin_amdgcn_s_wait_tensorcnt((short)0);
#endif
  unsigned long long gb =
      (unsigned long long)(void*)dst + (unsigned long long)r0 * D * 4ull;
  g0[2] = (unsigned)(gb & 0xffffffffu);
  g0[3] = (unsigned)((gb >> 32) & 0x01ffffffu) | (2u << 30);
#if defined(__clang_major__) && __clang_major__ >= 23
  __builtin_amdgcn_tensor_store_from_lds(g0, g1, g2, g3, gz, 0);
#else
  __builtin_amdgcn_tensor_store_from_lds(g0, g1, g2, g3, 0);
#endif
#if __has_builtin(__builtin_amdgcn_s_wait_tensorcnt)
  __builtin_amdgcn_s_wait_tensorcnt((short)0);
#endif
#else   /* fallback: plain vectorized copy */
  int r0 = blockIdx.x * TDM_ROWS;
  for (int i = threadIdx.x; i < TDM_ROWS * (D / 4); i += 32) {
    int r = r0 + i / (D / 4);
    if (r < rows)
      ((v4f*)dst)[(size_t)r * (D / 4) + (i % (D / 4))] =
          ((const v4f*)src)[(size_t)r * (D / 4) + (i % (D / 4))];
  }
#endif
}

/* ---------------- sequential eviction scan (age-offset trick) ---------------- */
__global__ __launch_bounds__(1024)
void evict_scan(const float* __restrict__ keys, const float* __restrict__ vals,
                const float* __restrict__ age_in, float* __restrict__ mk,
                float* __restrict__ mv, float* __restrict__ age)
{
  __shared__ float rv[1024];
  __shared__ int   ri[1024];
  __shared__ int   best;
  int t = threadIdx.x;
  for (int i = t; i < MEMROWS; i += 1024) age[i] = age_in[i];
  __syncthreads();
  for (int s = 0; s < Q; ++s) {
    float bv = -3.0e38f; int bi = 0;
    for (int i = t; i < MEMROWS; i += 1024) {
      float v = age[i];
      if (v > bv) { bv = v; bi = i; }
    }
    rv[t] = bv; ri[t] = bi;
    __syncthreads();
    for (int st = 512; st > 0; st >>= 1) {
      if (t < st) {
        if (rv[t + st] > rv[t] || (rv[t + st] == rv[t] && ri[t + st] < ri[t])) {
          rv[t] = rv[t + st]; ri[t] = ri[t + st];
        }
      }
      __syncthreads();
    }
    if (t == 0) { best = ri[0]; age[best] = -(float)s; }
    __syncthreads();
    if (t < D) {
      mk[(size_t)best * D + t] = keys[(size_t)s * D + t];
      mv[(size_t)best * D + t] = vals[(size_t)s * D + t];
    }
    __syncthreads();
  }
  for (int i = t; i < MEMROWS; i += 1024) age[i] += (float)Q;
}

/* =============================== host side =============================== */
extern "C" void kernel_launch(void* const* d_in, const int* in_sizes, int n_in,
                              void* d_out, int out_size, void* d_ws, size_t ws_size,
                              hipStream_t stream)
{
  (void)in_sizes; (void)n_in; (void)out_size; (void)ws_size;
  const float* x_in   = (const float*)d_in[0];
  const float* mkeys  = (const float*)d_in[1];
  const float* mvals  = (const float*)d_in[2];
  const float* age_in = (const float*)d_in[3];
  auto P = [&](int i) { return (const float*)d_in[i]; };

  float* ws = (float*)d_ws;
  size_t off = 0;
  float* xbuf  = ws + off; off += (size_t)Q * D;
  float* hbuf  = ws + off; off += (size_t)Q * D;
  float* obuf  = ws + off; off += (size_t)Q * D;
  float* keysb = ws + off; off += (size_t)Q * D;
  float* valsb = ws + off; off += (size_t)Q * D;
  float* qryb  = ws + off; off += (size_t)Q * D;
  float* qinv  = ws + off; off += Q;
  float* qkvb  = ws + off; off += (size_t)Q * 3 * D;
  float* mlpb  = ws + off; off += (size_t)Q * HID;
  float* minv  = ws + off; off += MEMROWS;
  float* pval  = ws + off; off += (size_t)NCHUNKS * Q * TOPK;
  int*   pidx  = (int*)(ws + off); off += (size_t)NCHUNKS * Q * TOPK;
  _Float16* qnh = (_Float16*)(ws + off); off += (size_t)Q * D / 2;

  hipMemcpyAsync(xbuf, x_in, sizeof(float) * (size_t)Q * D,
                 hipMemcpyDeviceToDevice, stream);

  /* -------- transformer blocks -------- */
  for (int l = 0; l < NLAYERS; ++l) {
    int base = 4 + l * 15;
    const float *ln1g = P(base + 0), *ln1b = P(base + 1);
    const float *qkvw = P(base + 2), *qkvbi = P(base + 3);
    const float *projw = P(base + 4), *projb = P(base + 5);
    const float *rel = P(base + 6);
    const float *ln2g = P(base + 7), *ln2b = P(base + 8);
    const float *fc1w = P(base + 9), *fc1b = P(base + 10);
    const float *fc2w = P(base + 11), *fc2b = P(base + 12);
    const float *g1 = P(base + 13), *g2 = P(base + 14);

    ln_rows<<<Q, 256, 0, stream>>>(xbuf, ln1g, ln1b, hbuf, D, 0);
    gemm_wmma<<<dim3(3 * D / 64, Q / 32), 256, 0, stream>>>(
        hbuf, qkvw, qkvbi, nullptr, nullptr, qkvb, Q, 3 * D, D, 0);
    attn_small<<<BATCH * NHEADS, 64, 0, stream>>>(qkvb, rel, obuf);
    gemm_wmma<<<dim3(D / 64, Q / 32), 256, 0, stream>>>(
        obuf, projw, projb, xbuf, g1, xbuf, Q, D, D, 0);
    ln_rows<<<Q, 256, 0, stream>>>(xbuf, ln2g, ln2b, hbuf, D, 0);
    gemm_wmma<<<dim3(HID / 64, Q / 32), 256, 0, stream>>>(
        hbuf, fc1w, fc1b, nullptr, nullptr, mlpb, Q, HID, D, 1);
    gemm_wmma<<<dim3(D / 64, Q / 32), 256, 0, stream>>>(
        mlpb, fc2w, fc2b, xbuf, g2, xbuf, Q, D, HID, 0);
  }

  /* -------- memory projections -------- */
  int mb = 4 + NLAYERS * 15;
  gemm_wmma<<<dim3(D / 64, Q / 32), 256, 0, stream>>>(
      xbuf, P(mb + 0), P(mb + 1), nullptr, nullptr, keysb, Q, D, D, 0);
  gemm_wmma<<<dim3(D / 64, Q / 32), 256, 0, stream>>>(
      xbuf, P(mb + 2), P(mb + 3), nullptr, nullptr, valsb, Q, D, D, 0);
  gemm_wmma<<<dim3(D / 64, Q / 32), 256, 0, stream>>>(
      xbuf, P(mb + 4), P(mb + 5), nullptr, nullptr, qryb, Q, D, D, 0);

  /* -------- fused KNN retrieval -------- */
  row_inv_norm<<<Q, 128, 0, stream>>>(qryb, qinv, D);
  scale_to_h<<<(Q * D + 255) / 256, 256, 0, stream>>>(qryb, qinv, qnh, Q * D);
  row_inv_norm<<<MEMROWS, 128, 0, stream>>>(mkeys, minv, D);
  knn_topk<<<NCHUNKS, 256, 0, stream>>>(qnh, mkeys, minv, pval, pidx);
  mem_attend<<<Q, 256, 0, stream>>>(pval, pidx, NCHUNKS, qryb, mkeys, mvals, xbuf);

  /* -------- prediction heads -------- */
  float* out_sp  = (float*)d_out;
  float* out_up  = out_sp + (size_t)Q * D;
  float* out_mk  = out_up + (size_t)Q * D;
  float* out_mv  = out_mk + (size_t)MEMROWS * D;
  float* out_age = out_mv + (size_t)MEMROWS * D;

  int sb = mb + 6;
  gemm_wmma<<<dim3(HID / 64, Q / 32), 256, 0, stream>>>(
      xbuf, P(sb + 0), P(sb + 1), nullptr, nullptr, mlpb, Q, HID, D, 0);
  ln_rows<<<Q, 256, 0, stream>>>(mlpb, P(sb + 2), P(sb + 3), mlpb, HID, 1);
  gemm_wmma<<<dim3(D / 64, Q / 32), 256, 0, stream>>>(
      mlpb, P(sb + 4), P(sb + 5), nullptr, nullptr, out_sp, Q, D, HID, 0);

  int ub = sb + 6;
  gemm_wmma<<<dim3(HID / 64, Q / 32), 256, 0, stream>>>(
      xbuf, P(ub + 0), P(ub + 1), nullptr, nullptr, mlpb, Q, HID, D, 0);
  ln_rows<<<Q, 256, 0, stream>>>(mlpb, P(ub + 2), P(ub + 3), mlpb, HID, 1);
  gemm_wmma<<<dim3(D / 64, Q / 32), 256, 0, stream>>>(
      mlpb, P(ub + 4), P(ub + 5), nullptr, nullptr, out_up, Q, D, HID, 2);

  /* -------- memory write-back via TDM + sequential eviction -------- */
  tdm_copy_rows<<<MEMROWS / TDM_ROWS, 32, 0, stream>>>(mkeys, out_mk, MEMROWS);
  tdm_copy_rows<<<MEMROWS / TDM_ROWS, 32, 0, stream>>>(mvals, out_mv, MEMROWS);
  evict_scan<<<1, 1024, 0, stream>>>(keysb, valsb, age_in, out_mk, out_mv, out_age);
}
